// Gemma3Synapse270M_84447646974509
// MI455X (gfx1250) — compile-verified, matrix operand order checked by reading, not run on hardware
//
#include <hip/hip_runtime.h>
#include <hip/hip_bf16.h>
#include <math.h>

// ---------------------------------------------------------------------------
// Gemma3-style forward pass for MI455X (gfx1250, wave32, WMMA).
// All matmuls: bf16 WMMA (v_wmma_f32_16x16x32_bf16), f32 accumulation.
// ---------------------------------------------------------------------------

typedef __bf16 bf16_t;
typedef __attribute__((ext_vector_type(8)))  __bf16 v8bf;
typedef __attribute__((ext_vector_type(16))) __bf16 v16bf;
typedef __attribute__((ext_vector_type(8)))  float  v8f;

#define D_MODEL 640
#define NHEADS  4
#define HEADDIM 256
#define QKV_Q   1024      // NH*HD
#define DFF_    2048
#define NLAYERS 18
#define SEQ     1024
#define VOCAB_  262144
#define EMB_SCALE 25.298221281347036f

// ---------------------------------------------------------------------------
// WMMA fragment loaders.
// 16-bit A/B fragment per-lane layout (ISA 7.12.2): lane L holds row (L&15);
// lanes 0-15 take K chunks [0,8) and [16,24), lanes 16-31 take [8,16),[24,32).
// Our "koff" = (lane>>4)<<3 is folded into the base pointer by the caller.
// ---------------------------------------------------------------------------
__device__ __forceinline__ v16bf make_frag(const bf16_t* p) {
  v8bf lo = *(const v8bf*)p;
  v8bf hi = *(const v8bf*)(p + 16);
  return __builtin_shufflevector(lo, hi, 0,1,2,3,4,5,6,7,8,9,10,11,12,13,14,15);
}
__device__ __forceinline__ v16bf make_frag(const float* p) {
  v16bf r;
#pragma unroll
  for (int i = 0; i < 8; ++i) { r[i] = (bf16_t)p[i]; r[i + 8] = (bf16_t)p[i + 16]; }
  return r;
}

// ---------------------------------------------------------------------------
// Generic NT GEMM: C[m, c_col0+n] (f32) = A_bf16[m, :K] . B[n, :K]
//   A: bf16, row stride lda.  B: f32 or bf16 weights, row stride ldb.
//   M is always SEQ=1024: 8 waves x 128 rows each; block owns 32 N columns.
//   Each weight element is read by exactly one block (streamed once from HBM).
//   MODE 1: scores epilogue (scale + causal sliding-window mask).
// ---------------------------------------------------------------------------
enum { MODE_PLAIN = 0, MODE_MASK = 1 };

template <typename TB, int MODE>
__global__ __launch_bounds__(256) void k_gemm_nt(
    const bf16_t* __restrict__ A, int lda,
    const TB* __restrict__ B, int ldb,
    float* __restrict__ C, int ldc, int c_col0,
    int K, float scale, int win)
{
  const int lane = threadIdx.x & 31;
  const int wave = threadIdx.x >> 5;
  const int n0   = blockIdx.x * 32;
  const int m0   = wave * 128;
  const int ln15 = lane & 15;
  const int koff = (lane >> 4) << 3;

  v8f zero;
#pragma unroll
  for (int e = 0; e < 8; ++e) zero[e] = 0.0f;

  v8f acc[8][2];
#pragma unroll
  for (int i = 0; i < 8; ++i) { acc[i][0] = zero; acc[i][1] = zero; }

  const TB*     bptr0 = B + (size_t)(n0 + ln15)      * ldb + koff;
  const TB*     bptr1 = B + (size_t)(n0 + 16 + ln15) * ldb + koff;
  const bf16_t* aptr  = A + (size_t)(m0 + ln15)      * lda + koff;

  for (int k0 = 0; k0 < K; k0 += 32) {
    if (k0 + 32 < K) {                       // gfx1250 global_prefetch_b8 path
      __builtin_prefetch(bptr0 + k0 + 32, 0, 0);
      __builtin_prefetch(bptr1 + k0 + 32, 0, 0);
    }
    v16bf b0 = make_frag(bptr0 + k0);
    v16bf b1 = make_frag(bptr1 + k0);
    const bf16_t* ap = aptr + k0;
#pragma unroll
    for (int mt = 0; mt < 8; ++mt) {
      v16bf a = make_frag(ap);
      ap += (size_t)16 * lda;
      acc[mt][0] = __builtin_amdgcn_wmma_f32_16x16x32_bf16(
          false, a, false, b0, (short)0, acc[mt][0], false, false);
      acc[mt][1] = __builtin_amdgcn_wmma_f32_16x16x32_bf16(
          false, a, false, b1, (short)0, acc[mt][1], false, false);
    }
  }

  // C/D layout (ISA 7.12.2): VGPR r -> M = r + (lane>=16 ? 8 : 0); N = lane&15.
  const int mhalf = (lane >> 4) << 3;
#pragma unroll
  for (int mt = 0; mt < 8; ++mt) {
#pragma unroll
    for (int nt = 0; nt < 2; ++nt) {
      const int col = n0 + nt * 16 + ln15;
#pragma unroll
      for (int r = 0; r < 8; ++r) {
        const int row = m0 + mt * 16 + mhalf + r;
        float v = acc[mt][nt][r];
        if (MODE == MODE_MASK) {
          v *= scale;
          if (col > row || col < row - win + 1) v = -3.0e38f;
        }
        C[(size_t)row * ldc + c_col0 + col] = v;
      }
    }
  }
}

// ---------------------------------------------------------------------------
// RMSNorm: y_bf16 = x * rsqrt(mean(x^2)+1e-6) * (1+w)
// ---------------------------------------------------------------------------
__global__ __launch_bounds__(256) void k_rms_bf16(
    const float* __restrict__ X, const float* __restrict__ W,
    bf16_t* __restrict__ Y, int Dd)
{
  __shared__ float red[256];
  const int t = blockIdx.x;
  const float* x = X + (size_t)t * Dd;
  float s = 0.f;
  for (int i = threadIdx.x; i < Dd; i += 256) { float v = x[i]; s += v * v; }
  red[threadIdx.x] = s; __syncthreads();
  for (int o = 128; o > 0; o >>= 1) {
    if (threadIdx.x < o) red[threadIdx.x] += red[threadIdx.x + o];
    __syncthreads();
  }
  const float inv = rsqrtf(red[0] / (float)Dd + 1e-6f);
  for (int i = threadIdx.x; i < Dd; i += 256)
    Y[(size_t)t * Dd + i] = (bf16_t)(x[i] * inv * (1.f + W[i]));
}

// O = R + rms(X)*(1+W)   (residual add, f32 out)
__global__ __launch_bounds__(256) void k_rms_add(
    const float* __restrict__ X, const float* __restrict__ W,
    const float* __restrict__ R, float* __restrict__ O, int Dd)
{
  __shared__ float red[256];
  const int t = blockIdx.x;
  const float* x = X + (size_t)t * Dd;
  float s = 0.f;
  for (int i = threadIdx.x; i < Dd; i += 256) { float v = x[i]; s += v * v; }
  red[threadIdx.x] = s; __syncthreads();
  for (int o = 128; o > 0; o >>= 1) {
    if (threadIdx.x < o) red[threadIdx.x] += red[threadIdx.x + o];
    __syncthreads();
  }
  const float inv = rsqrtf(red[0] / (float)Dd + 1e-6f);
  for (int i = threadIdx.x; i < Dd; i += 256)
    O[(size_t)t * Dd + i] = R[(size_t)t * Dd + i] + x[i] * inv * (1.f + W[i]);
}

// ---------------------------------------------------------------------------
// Per-head QK RMSNorm + RoPE -> bf16.  128 threads; thread i owns pair (i,i+128).
// ---------------------------------------------------------------------------
__global__ __launch_bounds__(128) void k_qknorm_rope(
    const float* __restrict__ X, int xStride,
    const float* __restrict__ W,
    bf16_t* __restrict__ Y, int yStride, float base)
{
  __shared__ float red[128];
  const int t = blockIdx.x, h = blockIdx.y, i = threadIdx.x;
  const float* x = X + (size_t)t * xStride + h * HEADDIM;
  bf16_t*      y = Y + (size_t)t * yStride + h * HEADDIM;
  const float u1 = x[i], u2 = x[i + 128];
  red[i] = u1 * u1 + u2 * u2; __syncthreads();
  for (int o = 64; o > 0; o >>= 1) { if (i < o) red[i] += red[i + o]; __syncthreads(); }
  const float inv = rsqrtf(red[0] / 256.f + 1e-6f);
  const float n1 = u1 * inv * (1.f + W[i]);
  const float n2 = u2 * inv * (1.f + W[i + 128]);
  const float ang = (float)t * powf(base, -(float)i / 128.f);
  float sn, cs; sincosf(ang, &sn, &cs);
  y[i]       = (bf16_t)(n1 * cs - n2 * sn);
  y[i + 128] = (bf16_t)(n1 * sn + n2 * cs);
}

// V (SEQ x HD, f32) -> V^T (HD x SEQ, bf16) so P.V maps onto the NT GEMM.
__global__ __launch_bounds__(256) void k_vT(
    const float* __restrict__ V, bf16_t* __restrict__ VT)
{
  const int idx = blockIdx.x * 256 + threadIdx.x;   // over SEQ*HEADDIM
  const int t = idx >> 8, d = idx & 255;
  VT[(size_t)d * SEQ + t] = (bf16_t)V[idx];
}

// Row softmax over SEQ scores -> bf16 probabilities (mask already -3e38).
__global__ __launch_bounds__(256) void k_softmax(
    const float* __restrict__ S, bf16_t* __restrict__ P)
{
  __shared__ float red[256];
  const int t = blockIdx.x;
  const float* s = S + (size_t)t * SEQ;
  float mx = -3.4e38f;
  for (int i = threadIdx.x; i < SEQ; i += 256) mx = fmaxf(mx, s[i]);
  red[threadIdx.x] = mx; __syncthreads();
  for (int o = 128; o > 0; o >>= 1) {
    if (threadIdx.x < o) red[threadIdx.x] = fmaxf(red[threadIdx.x], red[threadIdx.x + o]);
    __syncthreads();
  }
  mx = red[0]; __syncthreads();
  float sum = 0.f;
  for (int i = threadIdx.x; i < SEQ; i += 256) sum += __expf(s[i] - mx);
  red[threadIdx.x] = sum; __syncthreads();
  for (int o = 128; o > 0; o >>= 1) {
    if (threadIdx.x < o) red[threadIdx.x] += red[threadIdx.x + o];
    __syncthreads();
  }
  const float inv = 1.f / red[0];
  for (int i = threadIdx.x; i < SEQ; i += 256)
    P[(size_t)t * SEQ + i] = (bf16_t)(__expf(s[i] - mx) * inv);
}

// GeLU(g)*u -> bf16
__global__ __launch_bounds__(256) void k_gelu_mul(
    const float* __restrict__ G, const float* __restrict__ U,
    bf16_t* __restrict__ O, int n)
{
  const int i = blockIdx.x * 256 + threadIdx.x;
  if (i >= n) return;
  const float g = G[i];
  const float th = tanhf(0.7978845608028654f * (g + 0.044715f * g * g * g));
  O[i] = (bf16_t)(0.5f * g * (1.f + th) * U[i]);
}

__global__ __launch_bounds__(256) void k_cast_bf16(
    const float* __restrict__ X, bf16_t* __restrict__ Y, int n)
{
  const int i = blockIdx.x * 256 + threadIdx.x;
  if (i < n) Y[i] = (bf16_t)X[i];
}

__global__ __launch_bounds__(256) void k_embed(
    const int* __restrict__ ids, const float* __restrict__ E, float* __restrict__ X)
{
  const int t = blockIdx.x;
  const int id = ids[t];
  for (int i = threadIdx.x; i < D_MODEL; i += 256)
    X[(size_t)t * D_MODEL + i] = E[(size_t)id * D_MODEL + i] * EMB_SCALE;
}

// ---------------------------------------------------------------------------
// Host orchestration (all launches on `stream`; ws bump-allocated; graph-safe).
// ---------------------------------------------------------------------------
extern "C" void kernel_launch(void* const* d_in, const int* in_sizes, int n_in,
                              void* d_out, int out_size, void* d_ws, size_t ws_size,
                              hipStream_t stream) {
  (void)in_sizes; (void)n_in; (void)out_size; (void)ws_size;

  const int*   ids    = (const int*)  d_in[0];
  const float* embed  = (const float*)d_in[1];
  const float* ln_in  = (const float*)d_in[2];
  const float* wq     = (const float*)d_in[3];
  const float* wk     = (const float*)d_in[4];
  const float* wv     = (const float*)d_in[5];
  const float* qnw    = (const float*)d_in[6];
  const float* knw    = (const float*)d_in[7];
  const float* wo     = (const float*)d_in[8];
  const float* ln_pa  = (const float*)d_in[9];
  const float* ln_pf  = (const float*)d_in[10];
  const float* wg     = (const float*)d_in[11];
  const float* wu     = (const float*)d_in[12];
  const float* wd     = (const float*)d_in[13];
  const float* ln_ff  = (const float*)d_in[14];
  const float* norm_f = (const float*)d_in[15];
  float* logits = (float*)d_out;

  // workspace bump allocator
  char* wsb = (char*)d_ws;
  size_t off = 0;
  auto alloc = [&](size_t bytes) -> void* {
    void* p = wsb + off;
    off += (bytes + 255) & ~(size_t)255;
    return p;
  };
  float*  x    = (float*) alloc((size_t)SEQ * D_MODEL * 4);
  float*  x2   = (float*) alloc((size_t)SEQ * D_MODEL * 4);
  bf16_t* xn   = (bf16_t*)alloc((size_t)SEQ * D_MODEL * 2);
  float*  qf   = (float*) alloc((size_t)SEQ * QKV_Q   * 4);
  float*  kf   = (float*) alloc((size_t)SEQ * HEADDIM * 4);
  float*  vf   = (float*) alloc((size_t)SEQ * HEADDIM * 4);
  bf16_t* qr   = (bf16_t*)alloc((size_t)SEQ * QKV_Q   * 2);
  bf16_t* kr   = (bf16_t*)alloc((size_t)SEQ * HEADDIM * 2);
  bf16_t* vT   = (bf16_t*)alloc((size_t)HEADDIM * SEQ * 2);
  float*  sc   = (float*) alloc((size_t)SEQ * SEQ * 4);
  bf16_t* P    = (bf16_t*)alloc((size_t)SEQ * SEQ * 2);
  float*  ctxf = (float*) alloc((size_t)SEQ * QKV_Q * 4);
  bf16_t* ctxb = (bf16_t*)alloc((size_t)SEQ * QKV_Q * 2);
  float*  ao   = (float*) alloc((size_t)SEQ * D_MODEL * 4);
  float*  gbuf = (float*) alloc((size_t)SEQ * DFF_ * 4);
  float*  ubuf = (float*) alloc((size_t)SEQ * DFF_ * 4);
  bf16_t* gu   = (bf16_t*)alloc((size_t)SEQ * DFF_ * 2);

  k_embed<<<SEQ, 256, 0, stream>>>(ids, embed, x);

  for (int l = 0; l < NLAYERS; ++l) {
    const bool  glb  = ((l + 1) % 6) == 0;
    const float base = glb ? 1.0e6f : 1.0e4f;
    const int   win  = glb ? 32768 : 512;

    const float* wq_l = wq + (size_t)l * QKV_Q * D_MODEL;
    const float* wk_l = wk + (size_t)l * HEADDIM * D_MODEL;
    const float* wv_l = wv + (size_t)l * HEADDIM * D_MODEL;
    const float* wo_l = wo + (size_t)l * D_MODEL * QKV_Q;
    const float* wg_l = wg + (size_t)l * DFF_ * D_MODEL;
    const float* wu_l = wu + (size_t)l * DFF_ * D_MODEL;
    const float* wd_l = wd + (size_t)l * D_MODEL * DFF_;

    // pre-attention norm + QKV projections
    k_rms_bf16<<<SEQ, 256, 0, stream>>>(x, ln_in + (size_t)l * D_MODEL, xn, D_MODEL);
    k_gemm_nt<float, MODE_PLAIN><<<QKV_Q / 32, 256, 0, stream>>>(
        xn, D_MODEL, wq_l, D_MODEL, qf, QKV_Q, 0, D_MODEL, 1.f, 0);
    k_gemm_nt<float, MODE_PLAIN><<<HEADDIM / 32, 256, 0, stream>>>(
        xn, D_MODEL, wk_l, D_MODEL, kf, HEADDIM, 0, D_MODEL, 1.f, 0);
    k_gemm_nt<float, MODE_PLAIN><<<HEADDIM / 32, 256, 0, stream>>>(
        xn, D_MODEL, wv_l, D_MODEL, vf, HEADDIM, 0, D_MODEL, 1.f, 0);

    // QK norm + RoPE, V transpose
    k_qknorm_rope<<<dim3(SEQ, NHEADS), 128, 0, stream>>>(
        qf, QKV_Q, qnw + (size_t)l * HEADDIM, qr, QKV_Q, base);
    k_qknorm_rope<<<dim3(SEQ, 1), 128, 0, stream>>>(
        kf, HEADDIM, knw + (size_t)l * HEADDIM, kr, HEADDIM, base);
    k_vT<<<(SEQ * HEADDIM) / 256, 256, 0, stream>>>(vf, vT);

    // attention per head: QK^T (masked) -> softmax -> P.V
    for (int h = 0; h < NHEADS; ++h) {
      k_gemm_nt<bf16_t, MODE_MASK><<<SEQ / 32, 256, 0, stream>>>(
          qr + h * HEADDIM, QKV_Q, kr, HEADDIM, sc, SEQ, 0, HEADDIM, 0.0625f, win);
      k_softmax<<<SEQ, 256, 0, stream>>>(sc, P);
      k_gemm_nt<bf16_t, MODE_PLAIN><<<HEADDIM / 32, 256, 0, stream>>>(
          P, SEQ, vT, SEQ, ctxf, QKV_Q, h * HEADDIM, SEQ, 1.f, 0);
    }

    // output projection + post-attn norm + residual
    k_cast_bf16<<<(SEQ * QKV_Q) / 256, 256, 0, stream>>>(ctxf, ctxb, SEQ * QKV_Q);
    k_gemm_nt<float, MODE_PLAIN><<<D_MODEL / 32, 256, 0, stream>>>(
        ctxb, QKV_Q, wo_l, QKV_Q, ao, D_MODEL, 0, QKV_Q, 1.f, 0);
    k_rms_add<<<SEQ, 256, 0, stream>>>(ao, ln_pa + (size_t)l * D_MODEL, x, x2, D_MODEL);

    // MLP
    k_rms_bf16<<<SEQ, 256, 0, stream>>>(x2, ln_pf + (size_t)l * D_MODEL, xn, D_MODEL);
    k_gemm_nt<float, MODE_PLAIN><<<DFF_ / 32, 256, 0, stream>>>(
        xn, D_MODEL, wg_l, D_MODEL, gbuf, DFF_, 0, D_MODEL, 1.f, 0);
    k_gemm_nt<float, MODE_PLAIN><<<DFF_ / 32, 256, 0, stream>>>(
        xn, D_MODEL, wu_l, D_MODEL, ubuf, DFF_, 0, D_MODEL, 1.f, 0);
    k_gelu_mul<<<(SEQ * DFF_) / 256, 256, 0, stream>>>(gbuf, ubuf, gu, SEQ * DFF_);
    k_gemm_nt<float, MODE_PLAIN><<<D_MODEL / 32, 256, 0, stream>>>(
        gu, DFF_, wd_l, DFF_, ao, D_MODEL, 0, DFF_, 1.f, 0);
    k_rms_add<<<SEQ, 256, 0, stream>>>(ao, ln_ff + (size_t)l * D_MODEL, x2, x, D_MODEL);
  }

  // final norm + LM head (dominant GEMM: 1024 x 262144 x 640)
  k_rms_bf16<<<SEQ, 256, 0, stream>>>(x, norm_f, xn, D_MODEL);
  k_gemm_nt<float, MODE_PLAIN><<<VOCAB_ / 32, 256, 0, stream>>>(
      xn, D_MODEL, embed, D_MODEL, logits, VOCAB_, 0, D_MODEL, 1.f, 0);
}
